// RealLiquidTimeConstant_19756849562235
// MI455X (gfx1250) — matrix operand order repaired and checked by reading, not compiled, not run on hardware
//
#include <hip/hip_runtime.h>
#include <hip/hip_bf16.h>

#define IN_DIM   512
#define H_DIM    1024
#define OUT_DIM  256
#define B_DIM    64
#define S_DIM    256
#define UNFOLDS  6
#define EPS_F    1e-8f

#define NBLK     32
#define NTHREADS 256

typedef __attribute__((ext_vector_type(16))) __bf16 bf16x16;
typedef __attribute__((ext_vector_type(8)))  __bf16 bf16x8;
typedef __attribute__((ext_vector_type(8)))  float  f32x8;
typedef __attribute__((ext_vector_type(4)))  float  f32x4;

// ---------------------------------------------------------------------------
// Fragment helpers for V_WMMA_F32_16X16X32_BF16 (wave32).
// A (16xK, row-major): lane L holds row M=L&15; K contiguous 16 values at
//   kbase + khalf, khalf=(L>>4)*16.
// B (NxK row-major weight = B^T): lane L holds col N=n0+(L&15);
//   K contiguous 16 values at kbase + (L>>4)*16.
// ---------------------------------------------------------------------------
__device__ __forceinline__ bf16x16 load_a_bf16(const __bf16* __restrict__ rowp,
                                               int kbase, int khalf) {
    const __bf16* p = rowp + kbase + khalf;
    bf16x8 lo = *(const bf16x8*)(p);
    bf16x8 hi = *(const bf16x8*)(p + 8);
    bf16x16 a;
#pragma unroll
    for (int i = 0; i < 8; ++i) { a[i] = lo[i]; a[i + 8] = hi[i]; }
    return a;
}

__device__ __forceinline__ bf16x16 load_a_f32(const float* __restrict__ rowp,
                                              int kbase, int khalf) {
    const float* p = rowp + kbase + khalf;
    f32x4 p0 = *(const f32x4*)(p);
    f32x4 p1 = *(const f32x4*)(p + 4);
    f32x4 p2 = *(const f32x4*)(p + 8);
    f32x4 p3 = *(const f32x4*)(p + 12);
    bf16x16 a;
#pragma unroll
    for (int i = 0; i < 4; ++i) {
        a[i]      = (__bf16)p0[i];
        a[i + 4]  = (__bf16)p1[i];
        a[i + 8]  = (__bf16)p2[i];
        a[i + 12] = (__bf16)p3[i];
    }
    return a;
}

__device__ __forceinline__ bf16x16 load_b_glb(const __bf16* __restrict__ W,
                                              int K, int col, int kbase, int kb) {
    return *(const bf16x16*)(W + (size_t)col * K + kbase + kb);
}

// LDS B fragment: rows of RLEN bf16, XOR-swizzled in 16B chunks to kill the
// power-of-two row-stride bank conflict (chunk' = chunk ^ (row&15)).
__device__ __forceinline__ bf16x16 load_b_lds(const __bf16* sW, int jloc, int rlen,
                                              int kbase, int kb) {
    const int s  = jloc & 15;
    const int c0 = (kbase + kb) >> 3;
    const __bf16* base = sW + jloc * rlen;
    bf16x8 lo = *(const bf16x8*)(base + ((c0 ^ s) << 3));
    bf16x8 hi = *(const bf16x8*)(base + (((c0 + 1) ^ s) << 3));
    bf16x16 r;
#pragma unroll
    for (int i = 0; i < 8; ++i) { r[i] = lo[i]; r[i + 8] = hi[i]; }
    return r;
}

__device__ __forceinline__ float sigmoid_f(float z) {
    return 1.0f / (1.0f + __expf(-z));
}

// Sense-reversing grid barrier at AGENT (device) scope: coherence point is
// L2, so the release/acquire wb/inv only touch WGP$ (DEV-scope L2 wb/inv are
// no-ops) -- no L2 flush per barrier.
__device__ __forceinline__ void grid_sync(unsigned* cnt, unsigned* gen) {
    __syncthreads();
    if (threadIdx.x == 0) {
        unsigned g = __hip_atomic_load(gen, __ATOMIC_RELAXED, __HIP_MEMORY_SCOPE_AGENT);
        unsigned a = __hip_atomic_fetch_add(cnt, 1u, __ATOMIC_ACQ_REL,
                                            __HIP_MEMORY_SCOPE_AGENT) + 1u;
        if (a == (unsigned)NBLK) {
            __hip_atomic_store(cnt, 0u, __ATOMIC_RELAXED, __HIP_MEMORY_SCOPE_AGENT);
            __hip_atomic_store(gen, g + 1u, __ATOMIC_RELEASE, __HIP_MEMORY_SCOPE_AGENT);
        } else {
            while (__hip_atomic_load(gen, __ATOMIC_ACQUIRE, __HIP_MEMORY_SCOPE_AGENT) == g)
                __builtin_amdgcn_s_sleep(1);
        }
    }
    __syncthreads();
}

// ---------------------------------------------------------------------------
// Persistent kernel: entire (S=256) x (6 unfolds) recurrence in one launch.
// 32 blocks x 256 threads; each wave owns one fixed 16x16 (batch x hidden)
// tile for ALL phases, so i_in and the fp32 master copy of h live in
// registers; only the bf16 h (WMMA A-matrix, needed by all waves) goes
// through memory with ping-pong buffers + 6 grid barriers per step.
// W_rec and W_in slices are LDS-resident (swizzled) for the whole kernel.
// ---------------------------------------------------------------------------
__global__ void __launch_bounds__(NTHREADS)
k_persistent(const float* __restrict__ x, const __bf16* __restrict__ xb, int use_xb,
             const float* __restrict__ ts, const float* __restrict__ h0,
             const __bf16* __restrict__ win_b, const float* __restrict__ Winb,
             const __bf16* __restrict__ wrec_b,
             const __bf16* __restrict__ wout_b, const float* __restrict__ Woutb,
             const float* __restrict__ tau, const float* __restrict__ alpha,
             const float* __restrict__ beta, const float* __restrict__ mu,
             const float* __restrict__ sigma,
             __bf16* __restrict__ hbf_a, __bf16* __restrict__ hbf_b,
             float* __restrict__ y, float* __restrict__ hfin,
             unsigned* bar_cnt, unsigned* bar_gen) {
    __shared__ __bf16 sB[32 * H_DIM];                 // 64 KB w_rec slice
    __shared__ __bf16 sWin[32 * IN_DIM];              // 32 KB W_in slice

    const int blk   = blockIdx.x;
    const int w     = threadIdx.x >> 5;
    const int lane  = threadIdx.x & 31;
    const int nlo   = lane & 15;
    const int hi    = lane >> 4;
    const int khalf = hi * 16;
    const int kb    = hi * 16;
    const int m0    = 16 * (w & 3);
    const int n0    = blk * 32 + 16 * (w >> 2);       // this wave's H columns
    const int j     = n0 + nlo;                       // hidden unit (fixed)
    const int jloc  = j - blk * 32;                   // 0..31 within LDS slices
    const int arow  = m0 + nlo;                       // A-matrix row (batch)

    // Stage this block's 32 w_rec rows + 32 W_in rows into LDS (swizzled).
    {
        const __bf16* src = wrec_b + (size_t)blk * 32 * H_DIM;
        for (int q = threadIdx.x; q < 32 * 128; q += NTHREADS) {
            const int jj = q >> 7;
            const int c  = q & 127;
            bf16x8 v = *(const bf16x8*)(src + jj * H_DIM + c * 8);
            *(bf16x8*)(&sB[jj * H_DIM + ((c ^ (jj & 15)) << 3)]) = v;
        }
        const __bf16* srw = win_b + (size_t)blk * 32 * IN_DIM;
        for (int q = threadIdx.x; q < 32 * 64; q += NTHREADS) {
            const int jj = q >> 6;
            const int c  = q & 63;
            bf16x8 v = *(const bf16x8*)(srw + jj * IN_DIM + c * 8);
            *(bf16x8*)(&sWin[jj * IN_DIM + ((c ^ (jj & 15)) << 3)]) = v;
        }
    }
    __syncthreads();

    // Per-wave invariants.
    const float muj    = mu[j];
    const float isg    = __frcp_rn(sigma[j] + EPS_F);
    const float alj    = alpha[j];
    const float bej    = beta[j];
    const float ita    = __frcp_rn(tau[j]);
    const float bin_j  = Winb[j];
    const float bout_j = (blk < OUT_DIM / 32) ? Woutb[j] : 0.0f;
    const size_t arow_h = (size_t)arow * H_DIM;

    // fp32 master copy of this wave's 8 owned h elements, held in registers
    // for the whole sequence.
    float hreg[8];
#pragma unroll
    for (int r = 0; r < 8; ++r)
        hreg[r] = h0[(m0 + r + 8 * hi) * H_DIM + j];

    for (int t = 0; t < S_DIM; ++t) {
        // dt/ODE_UNFOLDS for this wave's 8 batches (overlaps phase A).
        float d_r[8];
#pragma unroll
        for (int r = 0; r < 8; ++r)
            d_r[r] = ts[(m0 + r + 8 * hi) * S_DIM + t] * (1.0f / (float)UNFOLDS);

        // -------- Phase A (wave-local, no barrier): i_in into registers ----
        float iin_r[8];
        {
            f32x8 acc = {};
            if (use_xb) {
                const __bf16* xrow = xb + ((size_t)arow * S_DIM + t) * IN_DIM;
                if (t + 1 < S_DIM)
                    __builtin_prefetch(xrow + IN_DIM, 0, 1);
#pragma unroll 4
                for (int k = 0; k < IN_DIM; k += 32) {
                    bf16x16 a = load_a_bf16(xrow, k, khalf);
                    bf16x16 b = load_b_lds(sWin, jloc, IN_DIM, k, kb);
                    acc = __builtin_amdgcn_wmma_f32_16x16x32_bf16(false, a, false, b,
                                                                  (short)0, acc, false, false);
                }
            } else {
                const float* xrow = x + ((size_t)arow * S_DIM + t) * IN_DIM;
                if (t + 1 < S_DIM)
                    __builtin_prefetch(xrow + IN_DIM, 0, 1);
#pragma unroll 2
                for (int k = 0; k < IN_DIM; k += 32) {
                    bf16x16 a = load_a_f32(xrow, k, khalf);
                    bf16x16 b = load_b_lds(sWin, jloc, IN_DIM, k, kb);
                    acc = __builtin_amdgcn_wmma_f32_16x16x32_bf16(false, a, false, b,
                                                                  (short)0, acc, false, false);
                }
            }
#pragma unroll
            for (int r = 0; r < 8; ++r) iin_r[r] = acc[r] + bin_j;
        }

        // -------- Phases U: 6 ODE unfolds ----------------------------------
        for (int u = 0; u < UNFOLDS; ++u) {
            const __bf16* bin  = (u & 1) ? hbf_b : hbf_a;
            __bf16*       bout = (u & 1) ? hbf_a : hbf_b;

            const __bf16* hrow = bin + arow_h;
            f32x8 acc = {};
#pragma unroll 4
            for (int k = 0; k < H_DIM; k += 32) {
                bf16x16 a = load_a_bf16(hrow, k, khalf);
                bf16x16 b = load_b_lds(sB, jloc, H_DIM, k, kb);
                acc = __builtin_amdgcn_wmma_f32_16x16x32_bf16(false, a, false, b,
                                                              (short)0, acc, false, false);
            }
#pragma unroll
            for (int r = 0; r < 8; ++r) {
                const float irec = acc[r];
                const float f    = sigmoid_f((irec - muj) * isg);
                const float hold = hreg[r];
                const float dhdt = -alj * hold + bej * f * (iin_r[r] + irec);
                const float hn   = hold + d_r[r] * dhdt * ita;
                hreg[r] = hn;
                bout[(m0 + r + 8 * hi) * H_DIM + j] = (__bf16)hn;
            }
            grid_sync(bar_cnt, bar_gen);
        }

        // -------- Phase O: y_t = h @ W_out^T + b_out (blocks 0..7) ---------
        // UNFOLDS even -> final bf16 state in hbf_a; overlaps next phase A.
        if (blk < OUT_DIM / 32) {
            const __bf16* hrow = hbf_a + arow_h;
            f32x8 acc = {};
#pragma unroll 2
            for (int k = 0; k < H_DIM; k += 32) {
                bf16x16 a = load_a_bf16(hrow, k, khalf);
                bf16x16 b = load_b_glb(wout_b, H_DIM, j, k, kb);
                acc = __builtin_amdgcn_wmma_f32_16x16x32_bf16(false, a, false, b,
                                                              (short)0, acc, false, false);
            }
#pragma unroll
            for (int r = 0; r < 8; ++r)
                y[((size_t)(m0 + r + 8 * hi) * S_DIM + t) * OUT_DIM + j] = acc[r] + bout_j;
        }
    }

    // h_final straight from the register master copy.
#pragma unroll
    for (int r = 0; r < 8; ++r)
        hfin[(m0 + r + 8 * hi) * H_DIM + j] = hreg[r];
}

// ---------------------------- prep kernels ---------------------------------
__global__ void k_cvt_mask(const float* __restrict__ w, const float* __restrict__ m,
                           __bf16* __restrict__ o, int n) {
    int i = blockIdx.x * 256 + threadIdx.x;
    if (i < n) o[i] = (__bf16)(w[i] * m[i]);
}
__global__ void k_cvt(const float* __restrict__ w, __bf16* __restrict__ o, int n) {
    int i = blockIdx.x * 256 + threadIdx.x;
    if (i < n) o[i] = (__bf16)w[i];
}
__global__ void k_init(const float* __restrict__ h0, __bf16* __restrict__ hb, int n,
                       unsigned* bar_cnt, unsigned* bar_gen) {
    int i = blockIdx.x * 256 + threadIdx.x;
    if (i < n) hb[i] = (__bf16)h0[i];
    if (i == 0) { *bar_cnt = 0u; *bar_gen = 0u; }
}

// ---------------------------------------------------------------------------
extern "C" void kernel_launch(void* const* d_in, const int* in_sizes, int n_in,
                              void* d_out, int out_size, void* d_ws, size_t ws_size,
                              hipStream_t stream) {
    (void)in_sizes; (void)n_in; (void)out_size;
    const float* x     = (const float*)d_in[0];
    const float* h0    = (const float*)d_in[1];
    const float* ts    = (const float*)d_in[2];
    const float* tau   = (const float*)d_in[3];
    const float* Winw  = (const float*)d_in[4];
    const float* Winb  = (const float*)d_in[5];
    const float* Wrecw = (const float*)d_in[6];
    const float* mask  = (const float*)d_in[7];
    const float* Woutw = (const float*)d_in[8];
    const float* Woutb = (const float*)d_in[9];
    const float* alpha = (const float*)d_in[10];
    const float* beta  = (const float*)d_in[11];
    const float* mu    = (const float*)d_in[12];
    const float* sigma = (const float*)d_in[13];
    float* out = (float*)d_out;

    char* ws = (char*)d_ws;
    size_t off = 0;
    auto alloc = [&](size_t bytes) -> void* {
        void* p = ws + off;
        off = (off + bytes + 255) & ~(size_t)255;
        return p;
    };
    __bf16*   wrec_b = (__bf16*)alloc((size_t)H_DIM * H_DIM * 2);
    __bf16*   win_b  = (__bf16*)alloc((size_t)H_DIM * IN_DIM * 2);
    __bf16*   wout_b = (__bf16*)alloc((size_t)OUT_DIM * H_DIM * 2);
    __bf16*   hbf_a  = (__bf16*)alloc((size_t)B_DIM * H_DIM * 2);
    __bf16*   hbf_b  = (__bf16*)alloc((size_t)B_DIM * H_DIM * 2);
    unsigned* bar    = (unsigned*)alloc(256);

    // Optional bf16 copy of x (removes fp32->bf16 converts from the critical
    // path).  Deterministic host-side choice based on fixed ws_size.
    const size_t xb_bytes = (size_t)B_DIM * S_DIM * IN_DIM * 2;
    __bf16* xb = nullptr;
    int use_xb = 0;
    if (off + xb_bytes <= ws_size) {
        xb = (__bf16*)alloc(xb_bytes);
        use_xb = 1;
    }

    k_cvt_mask<<<(H_DIM * H_DIM) / 256, 256, 0, stream>>>(Wrecw, mask, wrec_b, H_DIM * H_DIM);
    k_cvt<<<(H_DIM * IN_DIM) / 256, 256, 0, stream>>>(Winw, win_b, H_DIM * IN_DIM);
    k_cvt<<<(OUT_DIM * H_DIM) / 256, 256, 0, stream>>>(Woutw, wout_b, OUT_DIM * H_DIM);
    if (use_xb)
        k_cvt<<<(B_DIM * S_DIM * IN_DIM) / 256, 256, 0, stream>>>(x, xb,
                                                                  B_DIM * S_DIM * IN_DIM);
    k_init<<<(B_DIM * H_DIM) / 256, 256, 0, stream>>>(h0, hbf_a, B_DIM * H_DIM,
                                                      bar, bar + 1);

    k_persistent<<<NBLK, NTHREADS, 0, stream>>>(
        x, xb, use_xb, ts, h0,
        win_b, Winb, wrec_b, wout_b, Woutb,
        tau, alpha, beta, mu, sigma,
        hbf_a, hbf_b,
        out, out + (size_t)B_DIM * S_DIM * OUT_DIM,
        bar, bar + 1);
}
